// GraphJambaModel_39745627357333
// MI455X (gfx1250) — matrix operand order repaired and checked by reading, not compiled, not run on hardware
//
#include <hip/hip_runtime.h>

typedef __attribute__((ext_vector_type(16))) _Float16 v16h;
typedef __attribute__((ext_vector_type(8)))  _Float16 v8h;
typedef __attribute__((ext_vector_type(8)))  float    v8f;

// ---------------- problem constants ----------------
constexpr int Bsz = 8, Nn = 512, Fin = 32, Eg = 8192;
constexpr int H = 256, DI = 512, DC = 4, DTR = 16, DSt = 8;
constexpr int NH = 4, HD = 64, Ltot = Bsz * Nn;   // 4096
constexpr int OUTC = 1 + 4 + 3 + 512 + 512 + 512 + 8; // 1552
constexpr float EPSLN = 1e-5f;

// Fragment-swizzled LDS layout (wave32, ISA 7.12.2):
//   A 16x32 tile : element (m,k) -> lane = m + 16*((k>>3)&1),
//                  h = (k&7) | (((k>>4)&1)<<3); addr = lane*16 + h
//   B 32x16 tile : element (k,n) -> lane = n + 16*((k>>3)&1),
//                  h = (k&7) | (((k>>4)&1)<<3);                     addr = lane*16 + h
// => each lane's 16 f16 are contiguous (32B) -> 2x ds_load_b128 per fragment.

__device__ inline v16h frag_ld(const _Float16* base) {
  return *(const v16h*)(base + (threadIdx.x & 31) * 16);
}

__device__ inline v8h pack8(float a0,float a1,float a2,float a3,
                            float a4,float a5,float a6,float a7) {
  v8h v;
  v[0]=(_Float16)a0; v[1]=(_Float16)a1; v[2]=(_Float16)a2; v[3]=(_Float16)a3;
  v[4]=(_Float16)a4; v[5]=(_Float16)a5; v[6]=(_Float16)a6; v[7]=(_Float16)a7;
  return v;
}

// ---- stage A tile 64x32 (swizzled). FULL: no guards, float4 loads.
template <bool FULL>
__device__ inline void stage_A(const float* __restrict__ A, int lda, int M,
                               int K, int m0, int kt, _Float16* Asw) {
  const int tid = threadIdx.x;
#pragma unroll
  for (int o = tid; o < 256; o += 128) {
    int m = o >> 2, oct = o & 3;
    int gr = m0 + m;
    int kbeg = kt * 32 + oct * 8;
    v8h val;
    if (FULL) {
      const float4 f0 = *(const float4*)&A[(size_t)gr * lda + kbeg];
      const float4 f1 = *(const float4*)&A[(size_t)gr * lda + kbeg + 4];
      val = pack8(f0.x, f0.y, f0.z, f0.w, f1.x, f1.y, f1.z, f1.w);
    } else {
      int grc = gr < M ? gr : M - 1;
#pragma unroll
      for (int j = 0; j < 8; ++j) {
        int k = kbeg + j;
        int kc = k < K ? k : K - 1;
        float f = A[(size_t)grc * lda + kc];            // clamped, always legal
        val[j] = (gr < M && k < K) ? (_Float16)f : (_Float16)0.0f;
      }
    }
    int la = (m & 15) + ((oct & 1) << 4);
    int h0 = (oct >> 1) << 3;
    *(v8h*)&Asw[(m >> 4) * 512 + la * 16 + h0] = val;
  }
}

// ---- stage B tile 32x64 (swizzled). FULL: no guards.
template <bool FULL>
__device__ inline void stage_B(const float* __restrict__ Bw, int ldb, int N,
                               int K, int n0, int kt, _Float16* Bsw) {
  const int tid = threadIdx.x;
#pragma unroll
  for (int o = tid; o < 256; o += 128) {
    int s = o >> 6, idx = o & 63;
    int lb = idx & 31;
    int h0 = (idx >> 5) << 3;
    int gn = n0 + (s << 4) + (lb & 15);
    int kb = ((lb >> 4) << 3) + (h0 << 1);
    v8h val;
    if (FULL) {
#pragma unroll
      for (int j = 0; j < 8; ++j)
        val[j] = (_Float16)Bw[(size_t)(kt * 32 + kb + j) * ldb + gn];
    } else {
      int gnc = gn < N ? gn : N - 1;
#pragma unroll
      for (int j = 0; j < 8; ++j) {
        int k = kt * 32 + kb + j;
        int kc = k < K ? k : K - 1;
        float f = Bw[(size_t)kc * ldb + gnc];           // clamped, always legal
        val[j] = (gn < N && k < K) ? (_Float16)f : (_Float16)0.0f;
      }
    }
    *(v8h*)&Bsw[s * 512 + lb * 16 + h0] = val;
  }
}

// ================= generic tiled WMMA GEMM: C = act(A@B + bias) ===============
// A: MxK row-major (lda % 8 == 0), B: KxN row-major. Double-buffered LDS.
// act: 0 = none, 1 = relu, 2 = softplus   (bias applied iff bias != nullptr)
template <bool FULL>
__global__ __launch_bounds__(128)
void wmma_gemm_kernel(const float* __restrict__ A, int lda,
                      const float* __restrict__ Bw, int ldb,
                      const float* __restrict__ bias,
                      float* __restrict__ C, int ldc,
                      int M, int N, int K, int act) {
  __shared__ __align__(32) _Float16 Asw[2][4 * 512];
  __shared__ __align__(32) _Float16 Bsw[2][4 * 512];
  const int tid = threadIdx.x;
  const int w = tid >> 5, lane = tid & 31;
  const int m0 = blockIdx.y * 64, n0 = blockIdx.x * 64;

  v8f c[4];
  v8f zero = {};
#pragma unroll
  for (int n = 0; n < 4; ++n) c[n] = zero;

  const int kts = (K + 31) / 32;
  stage_A<FULL>(A, lda, M, K, m0, 0, Asw[0]);
  stage_B<FULL>(Bw, ldb, N, K, n0, 0, Bsw[0]);

  for (int kt = 0; kt < kts; ++kt) {
    __syncthreads();                    // buf[kt&1] staged; old reads complete
    int cur = kt & 1;
    if (kt + 1 < kts) {                 // prefetch next tile into other buffer
      stage_A<FULL>(A, lda, M, K, m0, kt + 1, Asw[cur ^ 1]);
      stage_B<FULL>(Bw, ldb, N, K, n0, kt + 1, Bsw[cur ^ 1]);
    }
    v16h af = frag_ld(&Asw[cur][w * 512]);
#pragma unroll
    for (int n = 0; n < 4; ++n) {
      v16h bf = frag_ld(&Bsw[cur][n * 512]);
      c[n] = __builtin_amdgcn_wmma_f32_16x16x32_f16(
          false, af, false, bf, (short)0, c[n], false, false);
    }
  }

  // epilogue: element (i) -> row = i + 8*(lane>>4), col = lane&15 (ISA C layout)
#pragma unroll
  for (int n = 0; n < 4; ++n) {
#pragma unroll
    for (int i = 0; i < 8; ++i) {
      int row = m0 + w * 16 + i + ((lane >> 4) << 3);
      int col = n0 + n * 16 + (lane & 15);
      if (FULL || (row < M && col < N)) {
        float v = c[n][i];
        if (bias) v += bias[col];
        if (act == 1) v = v > 0.0f ? v : 0.0f;
        else if (act == 2) v = v > 20.0f ? v : log1pf(__expf(v));
        C[row * ldc + col] = v;
      }
    }
  }
}

// ================= flash attention (wave32 WMMA, online softmax) ==============
// Q/K/V: (L, H) f32, head = blockIdx.y uses cols [head*HD, head*HD+HD). O: (L,H)
__device__ inline void stage_KV(const float* __restrict__ K,
                                const float* __restrict__ V, int hoff, int kt,
                                _Float16* Ksw, _Float16* Vsw) {
  const int tid = threadIdx.x;
  // K tile: B-swizzled for S = Q K^T (contraction = hd chunk, n = key);
  // contiguous hd reads -> float4 pairs.
#pragma unroll
  for (int o2 = tid; o2 < 256; o2 += 128) {
    int r = o2 >> 6, idx = o2 & 63;
    int lb = idx & 31;
    int h0 = (idx >> 5) << 3;
    int key = ((r & 1) << 4) + (lb & 15);
    int hdb = ((r >> 1) << 5) + ((lb >> 4) << 3) + (h0 << 1);
    const float* src = &K[(size_t)(kt * 32 + key) * H + hoff + hdb];
    const float4 f0 = *(const float4*)src;
    const float4 f1 = *(const float4*)(src + 4);
    *(v8h*)&Ksw[r * 512 + lb * 16 + h0] =
        pack8(f0.x, f0.y, f0.z, f0.w, f1.x, f1.y, f1.z, f1.w);
  }
  // V tile: B-swizzled for O += P V (contraction = key, n = hd); strided reads.
#pragma unroll
  for (int o2 = tid; o2 < 256; o2 += 128) {
    int s = o2 >> 6, idx = o2 & 63;
    int lb = idx & 31;
    int h0 = (idx >> 5) << 3;
    int n = (s << 4) + (lb & 15);
    int kb = ((lb >> 4) << 3) + (h0 << 1);
    v8h val;
#pragma unroll
    for (int j = 0; j < 8; ++j)
      val[j] = (_Float16)V[(size_t)(kt * 32 + kb + j) * H + hoff + n];
    *(v8h*)&Vsw[s * 512 + lb * 16 + h0] = val;
  }
}

__global__ __launch_bounds__(128)
void flash_attn_kernel(const float* __restrict__ Q, const float* __restrict__ K,
                       const float* __restrict__ V, float* __restrict__ O) {
  __shared__ __align__(32) _Float16 Ksw[2][4 * 512];
  __shared__ __align__(32) _Float16 Vsw[2][4 * 512];
  __shared__ __align__(32) _Float16 Ps[4][512];   // per-wave, wave-order DS only
  const int tid = threadIdx.x, w = tid >> 5, lane = tid & 31;
  const int q0 = blockIdx.x * 64;
  const int hoff = blockIdx.y * HD;

  // Q A-fragments (two 16x32 chunks over hd=64), pre-swizzled, scaled 1/sqrt(64)
  v16h qf[2];
  {
    int m = lane & 15, kb = (lane >> 4) << 3;
    const float* qrow = &Q[(size_t)(q0 + w * 16 + m) * H + hoff];
#pragma unroll
    for (int ch = 0; ch < 2; ++ch)
#pragma unroll
      for (int g = 0; g < 2; ++g) {
        int ks = ch * 32 + g * 16 + kb;
        const float4 f0 = *(const float4*)&qrow[ks];
        const float4 f1 = *(const float4*)&qrow[ks + 4];
        qf[ch][g * 8 + 0] = (_Float16)(f0.x * 0.125f);
        qf[ch][g * 8 + 1] = (_Float16)(f0.y * 0.125f);
        qf[ch][g * 8 + 2] = (_Float16)(f0.z * 0.125f);
        qf[ch][g * 8 + 3] = (_Float16)(f0.w * 0.125f);
        qf[ch][g * 8 + 4] = (_Float16)(f1.x * 0.125f);
        qf[ch][g * 8 + 5] = (_Float16)(f1.y * 0.125f);
        qf[ch][g * 8 + 6] = (_Float16)(f1.z * 0.125f);
        qf[ch][g * 8 + 7] = (_Float16)(f1.w * 0.125f);
      }
  }

  v8f zero = {};
  v8f o[4];
#pragma unroll
  for (int n = 0; n < 4; ++n) o[n] = zero;
  float mrow[8], lrow[8];
#pragma unroll
  for (int i = 0; i < 8; ++i) { mrow[i] = -1e30f; lrow[i] = 0.0f; }

  constexpr int NT = Ltot / 32;
  stage_KV(K, V, hoff, 0, Ksw[0], Vsw[0]);

  for (int kt = 0; kt < NT; ++kt) {
    __syncthreads();                    // tile kt staged; old tile reads done
    int cur = kt & 1;
    if (kt + 1 < NT) stage_KV(K, V, hoff, kt + 1, Ksw[cur ^ 1], Vsw[cur ^ 1]);

    // ---- S = Q K^T (two 16-key halves, hd reduced in two 32-chunks)
    v8f s[2];
#pragma unroll
    for (int half = 0; half < 2; ++half) {
      v8f acc = zero;
#pragma unroll
      for (int ch = 0; ch < 2; ++ch) {
        v16h bf = frag_ld(&Ksw[cur][(ch * 2 + half) * 512]);
        acc = __builtin_amdgcn_wmma_f32_16x16x32_f16(
            false, qf[ch], false, bf, (short)0, acc, false, false);
      }
      s[half] = acc;
    }

    // ---- online softmax over the 32-key tile; write P directly A-swizzled
#pragma unroll
    for (int i = 0; i < 8; ++i) {
      float s0 = s[0][i], s1 = s[1][i];
      float t = fmaxf(s0, s1);
#pragma unroll
      for (int msk = 1; msk < 16; msk <<= 1) t = fmaxf(t, __shfl_xor(t, msk, 32));
      float mn = fmaxf(mrow[i], t);
      float al = __expf(mrow[i] - mn);
      float p0 = __expf(s0 - mn), p1 = __expf(s1 - mn);
      float rs = p0 + p1;
#pragma unroll
      for (int msk = 1; msk < 16; msk <<= 1) rs += __shfl_xor(rs, msk, 32);
      lrow[i] = lrow[i] * al + rs;
      mrow[i] = mn;
      o[0][i] *= al; o[1][i] *= al; o[2][i] *= al; o[3][i] *= al;
      // P(m=prow, key): p0 at key=lane&15, p1 at key=16+(lane&15)
      int prow = i + ((lane >> 4) << 3);
      int la = prow + (((lane >> 3) & 1) << 4); // A-swizzle lane
      Ps[w][la * 16 + (lane & 7)] = (_Float16)p0;
      Ps[w][la * 16 + 8 + (lane & 7)] = (_Float16)p1;
    }

    // ---- O += P @ V  (Ps is wave-private; DS ops from one wave are in-order)
    v16h pf = frag_ld(&Ps[w][0]);
#pragma unroll
    for (int n = 0; n < 4; ++n) {
      v16h bf = frag_ld(&Vsw[cur][n * 512]);
      o[n] = __builtin_amdgcn_wmma_f32_16x16x32_f16(
          false, pf, false, bf, (short)0, o[n], false, false);
    }
  }

#pragma unroll
  for (int n = 0; n < 4; ++n)
#pragma unroll
    for (int i = 0; i < 8; ++i) {
      int row = q0 + w * 16 + i + ((lane >> 4) << 3);
      int col = hoff + n * 16 + (lane & 15);
      O[row * H + col] = o[n][i] / lrow[i];
    }
}

// ======================= small / elementwise kernels ==========================
__global__ void fill_kernel(float* p, float v, int n) {
  int i = blockIdx.x * blockDim.x + threadIdx.x;
  if (i < n) p[i] = v;
}

__global__ void deg_edge_kernel(const int* __restrict__ ei, float* deg) {
  int i = blockIdx.x * blockDim.x + threadIdx.x;
  if (i >= Bsz * Eg) return;
  int b = i / Eg, e = i % Eg;
  int dst = ei[b * 2 * Eg + Eg + e];
  atomicAdd(&deg[b * Nn + dst], 1.0f);
}

__global__ void dinv_kernel(const float* __restrict__ deg, float* dinv) {
  int i = blockIdx.x * blockDim.x + threadIdx.x;
  if (i < Ltot) dinv[i] = rsqrtf(deg[i]);
}

// one block per edge, 256 threads = features
__global__ void gcn_scatter_kernel(const int* __restrict__ ei,
                                   const float* __restrict__ msg,
                                   const float* __restrict__ dinv,
                                   float* __restrict__ gout) {
  int eidx = blockIdx.x;
  int b = eidx / Eg, e = eidx % Eg;
  int src = ei[b * 2 * Eg + e];
  int dst = ei[b * 2 * Eg + Eg + e];
  int gs = b * Nn + src, gd = b * Nn + dst;
  float nrm = dinv[gs] * dinv[gd];
  int f = threadIdx.x;
  atomicAdd(&gout[gd * H + f], msg[gs * H + f] * nrm);
}

// h = relu(gout + msg*dinv^2 + bias)
__global__ void gcn_finish_kernel(const float* __restrict__ gout,
                                  const float* __restrict__ msg,
                                  const float* __restrict__ dinv,
                                  const float* __restrict__ bias,
                                  float* __restrict__ hout) {
  int i = blockIdx.x * blockDim.x + threadIdx.x;
  if (i >= Ltot * H) return;
  int node = i / H, f = i % H;
  float v = gout[i] + msg[i] * dinv[node] * dinv[node] + bias[f];
  hout[i] = v > 0.0f ? v : 0.0f;
}

// depthwise causal conv over xz[:, :DI] -> silu -> xc
__global__ void conv_silu_kernel(const float* __restrict__ xz,
                                 const float* __restrict__ cW,
                                 const float* __restrict__ cb,
                                 float* __restrict__ xc) {
  int i = blockIdx.x * blockDim.x + threadIdx.x;
  if (i >= Ltot * DI) return;
  int l = i / DI, d = i % DI;
  float s = cb[d];
#pragma unroll
  for (int k = 0; k < DC; ++k) {
    int ll = l + k - (DC - 1);
    if (ll >= 0) s += xz[ll * (2 * DI) + d] * cW[d * DC + k];
  }
  xc[i] = s / (1.0f + __expf(-s));
}

// selective scan: one thread per channel d (512 independent recurrences)
__global__ void scan_kernel(const float* __restrict__ delta,
                            const float* __restrict__ proj,
                            const float* __restrict__ xc,
                            const float* __restrict__ A_log,
                            float* __restrict__ y) {
  int d = blockIdx.x * blockDim.x + threadIdx.x;
  if (d >= DI) return;
  float A[DSt], h[DSt];
#pragma unroll
  for (int s = 0; s < DSt; ++s) { A[s] = -__expf(A_log[d * DSt + s]); h[s] = 0.0f; }
  for (int l = 0; l < Ltot; ++l) {
    float dl = delta[l * DI + d];
    float xv = xc[l * DI + d];
    const float* pr = &proj[l * 32];
    float acc = 0.0f;
#pragma unroll
    for (int s = 0; s < DSt; ++s) {
      float dA = __expf(dl * A[s]);
      h[s] = dA * h[s] + dl * pr[DTR + s] * xv;
      acc += h[s] * pr[DTR + DSt + s];
    }
    y[l * DI + d] = acc;
  }
}

// y = (y + Dp*xc) * silu(zg)
__global__ void mamba_combine_kernel(float* __restrict__ y,
                                     const float* __restrict__ xc,
                                     const float* __restrict__ xz,
                                     const float* __restrict__ Dp) {
  int i = blockIdx.x * blockDim.x + threadIdx.x;
  if (i >= Ltot * DI) return;
  int l = i / DI, d = i % DI;
  float z = xz[l * (2 * DI) + DI + d];
  float sz = z / (1.0f + __expf(-z));
  y[i] = (y[i] + Dp[d] * xc[i]) * sz;
}

__global__ void fin_assemble_kernel(const float* __restrict__ h0,
                                    const float* __restrict__ m,
                                    const float* __restrict__ a,
                                    float* __restrict__ fin) {
  int i = blockIdx.x * blockDim.x + threadIdx.x;
  if (i >= Ltot * H) return;
  int l = i / H, f = i % H;
  fin[l * (3 * H) + f] = h0[i];
  fin[l * (3 * H) + H + f] = m[i];
  fin[l * (3 * H) + 2 * H + f] = a[i];
}

// res = fused + h ; layernorm
__global__ void ln_kernel(const float* __restrict__ fused,
                          const float* __restrict__ h0,
                          const float* __restrict__ g,
                          const float* __restrict__ be,
                          float* __restrict__ hf) {
  __shared__ float red[H];
  int l = blockIdx.x, t = threadIdx.x;
  float v = fused[l * H + t] + h0[l * H + t];
  red[t] = v; __syncthreads();
  for (int s = H / 2; s > 0; s >>= 1) { if (t < s) red[t] += red[t + s]; __syncthreads(); }
  float mu = red[0] / (float)H; __syncthreads();
  float dv = v - mu;
  red[t] = dv * dv; __syncthreads();
  for (int s = H / 2; s > 0; s >>= 1) { if (t < s) red[t] += red[t + s]; __syncthreads(); }
  float var = red[0] / (float)H;
  hf[l * H + t] = dv * rsqrtf(var + EPSLN) * g[t] + be[t];
}

__global__ void pool_kernel(const float* __restrict__ hf, float* __restrict__ pooled) {
  int i = blockIdx.x * blockDim.x + threadIdx.x;
  if (i >= Bsz * H) return;
  int b = i / H, f = i % H;
  float s = 0.0f;
  for (int n = 0; n < Nn; ++n) s += hf[(b * Nn + n) * H + f];
  pooled[i] = s / (float)Nn;
}

__global__ void heads_kernel(const float* __restrict__ pooled,
                             const float* cw, const float* cb,
                             const float* hw, const float* hb,
                             const float* tw, const float* tb,
                             const float* p1w, const float* p1b,
                             const float* p2w, const float* p2b,
                             const float* dw, const float* db,
                             const float* sw, const float* sb,
                             float* __restrict__ out) {
  int i = blockIdx.x * blockDim.x + threadIdx.x;
  if (i >= Bsz * OUTC) return;
  int b = i / OUTC, j = i % OUTC;
  const float* W; const float* bi; int col, nc;
  if (j < 1)        { W = cw;  bi = cb;  col = j;        nc = 1; }
  else if (j < 5)   { W = hw;  bi = hb;  col = j - 1;    nc = 4; }
  else if (j < 8)   { W = tw;  bi = tb;  col = j - 5;    nc = 3; }
  else if (j < 520) { W = p1w; bi = p1b; col = j - 8;    nc = 512; }
  else if (j < 1032){ W = p2w; bi = p2b; col = j - 520;  nc = 512; }
  else if (j < 1544){ W = dw;  bi = db;  col = j - 1032; nc = 512; }
  else              { W = sw;  bi = sb;  col = j - 1544; nc = 8; }
  float s = bi[col];
  for (int f = 0; f < H; ++f) s += pooled[b * H + f] * W[f * nc + col];
  out[i] = s;
}

// ============================== host launcher =================================
static inline void gemm(hipStream_t st, const float* A, int lda,
                        const float* Bw, int ldb, const float* bias,
                        float* C, int M, int N, int K, int act) {
  dim3 g((N + 63) / 64, (M + 63) / 64), b(128);
  bool full = (M % 64 == 0) && (N % 64 == 0) && (K % 32 == 0);
  if (full)
    wmma_gemm_kernel<true><<<g, b, 0, st>>>(A, lda, Bw, ldb, bias, C, N, M, N, K, act);
  else
    wmma_gemm_kernel<false><<<g, b, 0, st>>>(A, lda, Bw, ldb, bias, C, N, M, N, K, act);
}

static inline int ceildiv(int a, int b) { return (a + b - 1) / b; }

extern "C" void kernel_launch(void* const* d_in, const int* in_sizes, int n_in,
                              void* d_out, int out_size, void* d_ws, size_t ws_size,
                              hipStream_t stream) {
  const float* x      = (const float*)d_in[0];
  const int*   ei     = (const int*)d_in[1];
  const float* W_init = (const float*)d_in[2];
  const float* b_init = (const float*)d_in[3];
  const float* g1W = (const float*)d_in[4];  const float* g1b = (const float*)d_in[5];
  const float* g2W = (const float*)d_in[6];  const float* g2b = (const float*)d_in[7];
  const float* inW = (const float*)d_in[8];
  const float* cW  = (const float*)d_in[9];  const float* cb  = (const float*)d_in[10];
  const float* xpW = (const float*)d_in[11];
  const float* dtW = (const float*)d_in[12]; const float* dtb = (const float*)d_in[13];
  const float* A_log = (const float*)d_in[14];
  const float* Dp  = (const float*)d_in[15];
  const float* outW = (const float*)d_in[16];
  const float* Wq = (const float*)d_in[17]; const float* bq = (const float*)d_in[18];
  const float* Wk = (const float*)d_in[19]; const float* bk = (const float*)d_in[20];
  const float* Wv = (const float*)d_in[21]; const float* bv = (const float*)d_in[22];
  const float* Wo = (const float*)d_in[23]; const float* bo = (const float*)d_in[24];
  const float* fW = (const float*)d_in[25]; const float* fb = (const float*)d_in[26];
  const float* lng = (const float*)d_in[27]; const float* lnb = (const float*)d_in[28];

  // workspace layout (floats)
  float* ws = (float*)d_ws;
  size_t off = 0;
  auto alloc = [&](size_t n) { float* p = ws + off; off += n; return p; };
  float* h0    = alloc((size_t)Ltot * H);
  float* msg   = alloc((size_t)Ltot * H);
  float* gout  = alloc((size_t)Ltot * H);
  float* deg   = alloc(Ltot);
  float* dinv  = alloc(Ltot);
  float* xz    = alloc((size_t)Ltot * 2 * DI);
  float* xc    = alloc((size_t)Ltot * DI);
  float* proj  = alloc((size_t)Ltot * 32);
  float* delta = alloc((size_t)Ltot * DI);
  float* ybuf  = alloc((size_t)Ltot * DI);
  float* mout  = alloc((size_t)Ltot * H);
  float* vb2   = alloc((size_t)Ltot * H);
  float* ao    = alloc((size_t)Ltot * H);
  float* aout  = alloc((size_t)Ltot * H);
  float* fin   = alloc((size_t)Ltot * 3 * H);
  float* fused = alloc((size_t)Ltot * H);
  float* pooled = alloc((size_t)Bsz * H);
  float* qb = msg;   // reuse after GCN
  float* kb = gout;  // reuse after GCN
  float* hf = fin;   // reuse after fusion GEMM

  const int T = 256;

  // ---- init projection: h0 = relu(x @ W_init + b_init)
  gemm(stream, x, Fin, W_init, H, b_init, h0, Ltot, H, Fin, 1);

  // ---- degrees (self loop = 1) + dinv
  fill_kernel<<<ceildiv(Ltot, T), T, 0, stream>>>(deg, 1.0f, Ltot);
  deg_edge_kernel<<<ceildiv(Bsz * Eg, T), T, 0, stream>>>(ei, deg);
  dinv_kernel<<<ceildiv(Ltot, T), T, 0, stream>>>(deg, dinv);

  // ---- GCN layer 1
  gemm(stream, h0, H, g1W, H, nullptr, msg, Ltot, H, H, 0);
  fill_kernel<<<ceildiv(Ltot * H, T), T, 0, stream>>>(gout, 0.0f, Ltot * H);
  gcn_scatter_kernel<<<Bsz * Eg, H, 0, stream>>>(ei, msg, dinv, gout);
  gcn_finish_kernel<<<ceildiv(Ltot * H, T), T, 0, stream>>>(gout, msg, dinv, g1b, h0);

  // ---- GCN layer 2
  gemm(stream, h0, H, g2W, H, nullptr, msg, Ltot, H, H, 0);
  fill_kernel<<<ceildiv(Ltot * H, T), T, 0, stream>>>(gout, 0.0f, Ltot * H);
  gcn_scatter_kernel<<<Bsz * Eg, H, 0, stream>>>(ei, msg, dinv, gout);
  gcn_finish_kernel<<<ceildiv(Ltot * H, T), T, 0, stream>>>(gout, msg, dinv, g2b, h0);

  // ---- Mamba
  gemm(stream, h0, H, inW, 2 * DI, nullptr, xz, Ltot, 2 * DI, H, 0);
  conv_silu_kernel<<<ceildiv(Ltot * DI, T), T, 0, stream>>>(xz, cW, cb, xc);
  gemm(stream, xc, DI, xpW, DTR + 2 * DSt, nullptr, proj, Ltot, DTR + 2 * DSt, DI, 0);
  // delta = softplus(dt @ dt_W + dt_b), dt = proj[:, :16] (lda = 32)
  gemm(stream, proj, 32, dtW, DI, dtb, delta, Ltot, DI, DTR, 2);
  scan_kernel<<<ceildiv(DI, T), T, 0, stream>>>(delta, proj, xc, A_log, ybuf);
  mamba_combine_kernel<<<ceildiv(Ltot * DI, T), T, 0, stream>>>(ybuf, xc, xz, Dp);
  gemm(stream, ybuf, DI, outW, H, nullptr, mout, Ltot, H, DI, 0);

  // ---- MHA
  gemm(stream, h0, H, Wq, H, bq, qb, Ltot, H, H, 0);
  gemm(stream, h0, H, Wk, H, bk, kb, Ltot, H, H, 0);
  gemm(stream, h0, H, Wv, H, bv, vb2, Ltot, H, H, 0);
  flash_attn_kernel<<<dim3(Ltot / 64, NH), 128, 0, stream>>>(qb, kb, vb2, ao);
  gemm(stream, ao, H, Wo, H, bo, aout, Ltot, H, H, 0);

  // ---- fusion + LN + heads
  fin_assemble_kernel<<<ceildiv(Ltot * H, T), T, 0, stream>>>(h0, mout, aout, fin);
  gemm(stream, fin, 3 * H, fW, H, fb, fused, Ltot, H, 3 * H, 1);
  ln_kernel<<<Ltot, H, 0, stream>>>(fused, h0, lng, lnb, hf);
  pool_kernel<<<ceildiv(Bsz * H, T), T, 0, stream>>>(hf, pooled);
  heads_kernel<<<ceildiv(Bsz * OUTC, T), T, 0, stream>>>(
      pooled,
      (const float*)d_in[29], (const float*)d_in[30],
      (const float*)d_in[31], (const float*)d_in[32],
      (const float*)d_in[33], (const float*)d_in[34],
      (const float*)d_in[35], (const float*)d_in[36],
      (const float*)d_in[37], (const float*)d_in[38],
      (const float*)d_in[39], (const float*)d_in[40],
      (const float*)d_in[41], (const float*)d_in[42],
      (float*)d_out);
}
